// AttentionHead_37898791420154
// MI455X (gfx1250) — matrix operand order repaired and checked by reading, not compiled, not run on hardware
//
#include <hip/hip_runtime.h>
#include <hip/hip_bf16.h>

// Problem constants (match reference)
#define EMBED 1024
#define HD    64
#define BATCH 8
#define SEQ   2048
#define KSPLIT 4
#define KSEG  (SEQ / KSPLIT)  // 512 keys per wave

typedef __attribute__((ext_vector_type(16))) __bf16 v16bf;
typedef __attribute__((ext_vector_type(8)))  __bf16 v8bf;
typedef __attribute__((ext_vector_type(8)))  float  v8f;

static __device__ __forceinline__ v8f wmma_bf16(v16bf a, v16bf b, v8f c) {
  // D = A(16x32 bf16) * B(32x16 bf16) + C(16x16 f32)
  return __builtin_amdgcn_wmma_f32_16x16x32_bf16(
      /*neg_a=*/false, a, /*neg_b=*/false, b,
      /*c_mod=*/(short)0, c, /*reuse_a=*/false, /*reuse_b=*/false);
}

static __device__ __forceinline__ v16bf cat8(v8bf a, v8bf b) {
  return __builtin_shufflevector(a, b, 0,1,2,3,4,5,6,7,8,9,10,11,12,13,14,15);
}

// ---------------------------------------------------------------------------
// Kernel 0: one-time weight prep.  W[1024x64] fp32 -> Wt[64x1024] bf16
// (transposed so projection B-fragments are contiguous 32B loads).
// ---------------------------------------------------------------------------
__global__ __launch_bounds__(256) void prep_w(
    const float* __restrict__ Wq, const float* __restrict__ Wk,
    const float* __restrict__ Wv, __bf16* __restrict__ Wt) {
  const int which = blockIdx.y;
  const float* W = (which == 0) ? Wq : (which == 1) ? Wk : Wv;
  __bf16* O = Wt + (size_t)which * EMBED * HD;
  const int t  = blockIdx.x * 256 + threadIdx.x;  // 0..65535
  const int kk = t >> 6;                          // embed index
  const int nn = t & 63;                          // head-dim index
  O[(size_t)nn * EMBED + kk] = (__bf16)W[t];      // coalesced read
}

// ---------------------------------------------------------------------------
// Kernel 1: fused QKV projection, no LDS, no barriers.
// P = X[16384x1024] * W[1024x64] + bias.  q/k stored bf16 row-major [M x 64];
// v stored bf16 TRANSPOSED per batch [8][64][2048].  blockIdx.y selects q/k/v.
// ---------------------------------------------------------------------------
__global__ __launch_bounds__(256) void qkv_proj(
    const float* __restrict__ Q, const float* __restrict__ K,
    const float* __restrict__ V, const __bf16* __restrict__ Wt,
    const float* __restrict__ bq, const float* __restrict__ bk,
    const float* __restrict__ bv,
    __bf16* __restrict__ qo, __bf16* __restrict__ ko,
    __bf16* __restrict__ vt) {
  const int which = blockIdx.y;
  const float* X    = (which == 0) ? Q  : (which == 1) ? K  : V;
  const float* bias = (which == 0) ? bq : (which == 1) ? bk : bv;
  const __bf16* Wm  = Wt + (size_t)which * EMBED * HD;  // [64][1024] bf16

  const int tid  = threadIdx.x;
  const int wave = tid >> 5;
  const int lane = tid & 31;
  const int l16  = lane & 15;
  const int hi   = lane >> 4;
  const int rowBase = blockIdx.x * 128 + wave * 16;
  const int arow    = rowBase + l16;   // A-fragment row for this lane
  const int kbase   = hi ? 16 : 0;     // K base per lane half (contiguous k = 16h + s)

  v8f acc[4] = {};  // 16x64 output tile = 4 N-tiles of 16

  for (int kc = 0; kc < EMBED; kc += 32) {
    // A fragment (16x32 bf16): elements 0..15 -> K=kbase+0..15 (contiguous).
    v16bf a;
    {
      const float* xp = X + (size_t)arow * EMBED + kc + kbase;
#pragma unroll
      for (int j = 0; j < 8; ++j) a[j] = (__bf16)xp[j];
#pragma unroll
      for (int j = 0; j < 8; ++j) a[8 + j] = (__bf16)xp[8 + j];
    }
    // B fragments: lane n -> col nt*16+l16, contiguous K run from Wt (L2-hot).
#pragma unroll
    for (int nt = 0; nt < 4; ++nt) {
      const __bf16* bp =
          Wm + (size_t)(nt * 16 + l16) * EMBED + kc + (hi ? 16 : 0);
      v16bf b = *(const v16bf*)bp;  // contiguous 32B load
      acc[nt] = wmma_bf16(a, b, acc[nt]);
    }
  }

  // C/D layout: VGPR r -> row r (lanes 0-15) / row 8+r (lanes 16-31),
  // col = l16 within the N-tile.
  if (which == 2) {
    // Store V transposed: vt[batch][col][seq]; each lane's 8 rows are
    // consecutive seq positions -> one 16B store per N-tile.
    const int b     = rowBase >> 11;               // / SEQ
    const int sBase = (rowBase & (SEQ - 1)) + hi * 8;
#pragma unroll
    for (int nt = 0; nt < 4; ++nt) {
      const int col = nt * 16 + l16;
      const float bb = bias[col];
      v8bf o;
#pragma unroll
      for (int r = 0; r < 8; ++r) o[r] = (__bf16)(acc[nt][r] + bb);
      *(v8bf*)(vt + (size_t)b * HD * SEQ + (size_t)col * SEQ + sBase) = o;
    }
  } else {
    __bf16* O = (which == 0) ? qo : ko;
#pragma unroll
    for (int nt = 0; nt < 4; ++nt) {
      const int col = nt * 16 + l16;
      const float bb = bias[col];
#pragma unroll
      for (int r = 0; r < 8; ++r) {
        const int row = rowBase + r + hi * 8;
        O[(size_t)row * HD + col] = (__bf16)(acc[nt][r] + bb);
      }
    }
  }
}

// ---------------------------------------------------------------------------
// Kernel 2: streaming (flash) attention with 4-way key split per row-group.
// Block = 256 threads = 8 waves = 2 row-groups x 4 key-quarters.
// Each wave owns 16 query rows and streams 512 keys (chunks of 64) with an
// online softmax; partials (m, l, acc) are merged exactly in LDS at the end.
// Grid = (64, 8) -> 512 workgroups / 4096 waves for latency hiding.
// ---------------------------------------------------------------------------
__global__ __launch_bounds__(256) void attn(
    const __bf16* __restrict__ q, const __bf16* __restrict__ k,
    const __bf16* __restrict__ vt, float* __restrict__ out) {
  __shared__ __bf16 Plds[8][16 * 64];   // per-wave P tile (16 q x 64 keys)
  __shared__ float  AccLds[8][16 * 64]; // per-wave partial output (f32)
  __shared__ float  MLds[8][16][2];     // per-wave per-row (m, l)

  const int tid  = threadIdx.x;
  const int wave = tid >> 5;
  const int lane = tid & 31;
  const int l16  = lane & 15;
  const int hi   = lane >> 4;
  const int g    = wave >> 2;  // row-group within block (0..1)
  const int qd   = wave & 3;   // key quarter (0..3)
  const int batch = blockIdx.y;
  const int qBase = blockIdx.x * 32 + g * 16;
  const int kBeg = qd * KSEG;
  const int kEnd = kBeg + KSEG;
  const size_t rOff = (size_t)batch * SEQ * HD;  // q/k row-major base
  const size_t vOff = (size_t)batch * HD * SEQ;  // vt [64][2048] base
  __bf16* P = &Plds[wave][0];

  // Hoist Q A-fragments (head dim split into two K=32 steps).
  const int arow = qBase + l16;
  v16bf qa[2];
  {
    const __bf16* qp = q + rOff + (size_t)arow * HD;
#pragma unroll
    for (int ks = 0; ks < 2; ++ks) {
      const __bf16* p = qp + ks * 32 + (hi ? 16 : 0);
      qa[ks] = cat8(*(const v8bf*)p, *(const v8bf*)(p + 8));
    }
  }

  float m[8], ll[8];
#pragma unroll
  for (int r = 0; r < 8; ++r) { m[r] = -1e30f; ll[r] = 0.0f; }
  v8f acc[4] = {};
  const float scale = 0.125f;  // 1/sqrt(64)

  for (int kc = kBeg; kc < kEnd; kc += 64) {
    // ---- scores: 4 tiles of 16 keys, K=64 head dims in two WMMA steps ----
    v8f c[4] = {};
#pragma unroll
    for (int nt2 = 0; nt2 < 4; ++nt2) {
      // B fragment: lane n -> key row kc+nt2*16+l16, contiguous 16 head dims.
      const __bf16* kp =
          k + rOff + (size_t)(kc + nt2 * 16 + l16) * HD + (hi ? 16 : 0);
      v16bf b0 = *(const v16bf*)kp;         // head dims 0..31 half
      v16bf b1 = *(const v16bf*)(kp + 32);  // head dims 32..63 half
      c[nt2] = wmma_bf16(qa[0], b0, c[nt2]);
      c[nt2] = wmma_bf16(qa[1], b1, c[nt2]);
    }

    // Prefetch next chunk's K rows and V columns (global_prefetch_b8).
    if (kc + 64 < kEnd) {
      __builtin_prefetch(k + rOff + (size_t)(kc + 64 + lane) * HD, 0, 1);
      __builtin_prefetch(vt + vOff + (size_t)lane * SEQ + kc + 64, 0, 1);
    }

    // ---- online softmax across 64 keys (16-lane row-group reductions) ----
    float sc[8];
#pragma unroll
    for (int r = 0; r < 8; ++r) {
      const float s0 = c[0][r] * scale;
      const float s1 = c[1][r] * scale;
      const float s2 = c[2][r] * scale;
      const float s3 = c[3][r] * scale;
      float mx = fmaxf(fmaxf(s0, s1), fmaxf(s2, s3));
#pragma unroll
      for (int d = 1; d < 16; d <<= 1) mx = fmaxf(mx, __shfl_xor(mx, d));
      const float mn = fmaxf(m[r], mx);
      const float e0 = __expf(s0 - mn);
      const float e1 = __expf(s1 - mn);
      const float e2 = __expf(s2 - mn);
      const float e3 = __expf(s3 - mn);
      float sum = (e0 + e1) + (e2 + e3);
#pragma unroll
      for (int d = 1; d < 16; d <<= 1) sum += __shfl_xor(sum, d);
      const float rsc = __expf(m[r] - mn);
      ll[r] = ll[r] * rsc + sum;
      m[r]  = mn;
      sc[r] = rsc;
      // Park P in per-wave LDS (row-major [16 q][64 keys], bf16).
      const int prow = r + hi * 8;
      P[prow * 64 + l16]      = (__bf16)e0;
      P[prow * 64 + 16 + l16] = (__bf16)e1;
      P[prow * 64 + 32 + l16] = (__bf16)e2;
      P[prow * 64 + 48 + l16] = (__bf16)e3;
    }

    // Rescale running output accumulators.
#pragma unroll
    for (int nt = 0; nt < 4; ++nt) {
#pragma unroll
      for (int r = 0; r < 8; ++r) acc[nt][r] *= sc[r];
    }

    __builtin_amdgcn_wave_barrier();  // keep DS store->load program order

    // ---- acc += P(16x64) * V(64x64): two K=32 steps x 4 N-tiles ----
#pragma unroll
    for (int ks = 0; ks < 2; ++ks) {
      const __bf16* pp = P + l16 * 64 + ks * 32 + (hi ? 16 : 0);
      v16bf pa = cat8(*(const v8bf*)pp, *(const v8bf*)(pp + 8));
#pragma unroll
      for (int nt = 0; nt < 4; ++nt) {
        // B fragment from transposed V: contiguous 16 keys per lane.
        const __bf16* vbp = vt + vOff + (size_t)(nt * 16 + l16) * SEQ + kc +
                            ks * 32 + (hi ? 16 : 0);
        v16bf b = *(const v16bf*)vbp;
        acc[nt] = wmma_bf16(pa, b, acc[nt]);
      }
    }
    __builtin_amdgcn_wave_barrier();
  }

  // ---- publish this wave's partial (m, l, unnormalized acc) to LDS ----
#pragma unroll
  for (int nt = 0; nt < 4; ++nt) {
#pragma unroll
    for (int r = 0; r < 8; ++r) {
      AccLds[wave][(r + hi * 8) * 64 + nt * 16 + l16] = acc[nt][r];
    }
  }
  if (l16 == 0) {
#pragma unroll
    for (int r = 0; r < 8; ++r) {
      MLds[wave][r + hi * 8][0] = m[r];
      MLds[wave][r + hi * 8][1] = ll[r];
    }
  }
  __syncthreads();

  // ---- quarter-0 wave of each row-group merges the 4 partials exactly ----
  if (qd == 0) {
#pragma unroll
    for (int r = 0; r < 8; ++r) {
      const int row = r + hi * 8;
      float M = MLds[wave][row][0];
#pragma unroll
      for (int i = 1; i < KSPLIT; ++i) M = fmaxf(M, MLds[wave + i][row][0]);
      float scl[KSPLIT];
      float L = 0.0f;
#pragma unroll
      for (int i = 0; i < KSPLIT; ++i) {
        const float e = __expf(MLds[wave + i][row][0] - M);
        scl[i] = e;
        L += MLds[wave + i][row][1] * e;
      }
      const float invL = 1.0f / L;
      const int orow = qBase + row;
#pragma unroll
      for (int nt = 0; nt < 4; ++nt) {
        const int col = nt * 16 + l16;
        float s = 0.0f;
#pragma unroll
        for (int i = 0; i < KSPLIT; ++i) {
          s += AccLds[wave + i][row * 64 + col] * scl[i];
        }
        out[rOff + (size_t)orow * HD + col] = s * invL;
      }
    }
  }
}

// ---------------------------------------------------------------------------
extern "C" void kernel_launch(void* const* d_in, const int* in_sizes, int n_in,
                              void* d_out, int out_size, void* d_ws,
                              size_t ws_size, hipStream_t stream) {
  (void)in_sizes; (void)n_in; (void)out_size; (void)ws_size;
  const float* query = (const float*)d_in[0];
  const float* key_  = (const float*)d_in[1];
  const float* value = (const float*)d_in[2];
  const float* Wq    = (const float*)d_in[3];
  const float* bq    = (const float*)d_in[4];
  const float* Wk    = (const float*)d_in[5];
  const float* bk    = (const float*)d_in[6];
  const float* Wv    = (const float*)d_in[7];
  const float* bv    = (const float*)d_in[8];
  float* out = (float*)d_out;

  const int M = BATCH * SEQ;  // 16384 rows total
  __bf16* qb  = (__bf16*)d_ws;                   // [M x 64] bf16 row-major
  __bf16* kb  = qb + (size_t)M * HD;             // [M x 64] bf16 row-major
  __bf16* vtb = kb + (size_t)M * HD;             // [8][64][2048] bf16
  __bf16* Wt  = vtb + (size_t)M * HD;            // [3][64][1024] bf16

  dim3 g0(EMBED * HD / 256, 3);
  prep_w<<<g0, 256, 0, stream>>>(Wq, Wk, Wv, Wt);

  dim3 g1(M / 128, 3);  // 128 row-tiles x {q,k,v}
  qkv_proj<<<g1, 256, 0, stream>>>(query, key_, value, Wt, bq, bk, bv,
                                   qb, kb, vtb);

  dim3 g2(SEQ / 32, BATCH);  // 64 blocks x 8 batches; 2 row-groups x 4 splits
  attn<<<g2, 256, 0, stream>>>(qb, kb, vtb, out);
}
